// PupilTrackerModel_9723805958442
// MI455X (gfx1250) — matrix operand order repaired and checked by reading, not compiled, not run on hardware
//
#include <hip/hip_runtime.h>
#include <hip/hip_bf16.h>

typedef _Float16 half_t;
typedef __attribute__((ext_vector_type(16))) _Float16 v16h;
typedef __attribute__((ext_vector_type(8)))  float    v8f;

#define B_ 8
#define T_ 40

// ---------------- utility kernels ----------------
__global__ void fill_zero_f32(float* p, int n) {
    int i = blockIdx.x * blockDim.x + threadIdx.x;
    if (i < n) p[i] = 0.f;
}
__global__ void cast_f32_to_f16(const float* __restrict__ s, half_t* __restrict__ d, int n) {
    int i = blockIdx.x * blockDim.x + threadIdx.x;
    if (i < n) d[i] = (half_t)s[i];
}

// ---------------- pack weights into WMMA A-fragment order ----------------
// A 16-bit 16x32 per-lane layout (contiguous k order, matches B fills):
//   lanes 0-15 : M=lane,   element i -> K=i (0..15)
//   lanes16-31 : M=lane-16,element i -> K=16+i (16..31)
// dst layout: frag = (mb*Taps + tap)*KB + kb ; 32 lanes x 16 halfs contiguous.
__global__ void pack_a_kernel(const float* __restrict__ src, half_t* __restrict__ dst,
                              int MB, int Taps, int KB, int Ksrc,
                              int row_stride, int k_stride)
{
    int idx  = blockIdx.x * blockDim.x + threadIdx.x;
    int lane = idx & 31;
    int frag = idx >> 5;
    if (frag >= MB * Taps * KB) return;
    int kb  = frag % KB;
    int t2  = frag / KB;
    int tap = t2 % Taps;
    int mb  = t2 / Taps;
    int m   = mb * 16 + (lane & 15);
    bool lo = lane < 16;
    half_t* o = dst + ((long)frag * 32 + lane) * 16;
#pragma unroll
    for (int i = 0; i < 16; ++i) {
        int kk = (lo ? 0 : 16) + i;
        int k = kb * 32 + kk;
        float v = (k < Ksrc) ? src[(long)m * row_stride + (long)k * k_stride + tap] : 0.f;
        o[i] = (half_t)v;
    }
}

// ---------------- build channel-last spatially-padded input buffer ----------------
// inbuf[t][b][H+2][W+2][Kcpad] : interior channels 0..Cin-1 = x_t, everything else
// zero (halo ring, pad channels, and h-part of slot 0 == h0; later slots' interior
// h-part is overwritten by lstm_gate before being read).
__global__ void init_inbuf(const half_t* __restrict__ xsrc, half_t* __restrict__ inbuf,
                           int Cin, int Kcpad, int H, int W, int x_bstride, long n)
{
    long i = (long)blockIdx.x * blockDim.x + threadIdx.x;
    if (i >= n) return;
    int Wp = W + 2, Hp = H + 2;
    int c  = (int)(i % Kcpad); long r = i / Kcpad;
    int xw = (int)(r % Wp);    r /= Wp;
    int yh = (int)(r % Hp);    r /= Hp;
    int b  = (int)(r % B_);
    int t  = (int)(r / B_);
    int y = yh - 1, x = xw - 1;
    half_t v = (half_t)0.f;
    if (c < Cin && y >= 0 && y < H && x >= 0 && x < W)
        v = xsrc[(long)b * x_bstride + (long)t * Cin * H * W + (long)c * H * W + y * W + x];
    inbuf[i] = v;
}

// ---------------- ConvLSTM gate pre-activation conv (implicit GEMM, WMMA) ----------------
// z[b][4*Hid][H][W] = conv3x3(channel-last spatially padded input) + bias
// Block: 128 threads = 4 waves; each wave: M=16 out-channels x N=16 pixels.
// Fully unrolled 9*KB K-loop; all loads are base+constant-immediate b128s; no
// predication (halo is physically zero-padded).
template<int KB, int H, int W>
__global__ __launch_bounds__(128)
void conv_gates_wmma(const half_t* __restrict__ inslot, // [b][H+2][W+2][Kcpad]
                     const half_t* __restrict__ wpack,
                     const float* __restrict__ bias,
                     float* __restrict__ z, int Hid)
{
    constexpr int Kcpad = KB * 32;
    constexpr int Wp = W + 2, Hp = H + 2;
    constexpr int HW = H * W;
    const int tid  = threadIdx.x;
    const int wave = tid >> 5, lane = tid & 31;
    const int mb   = blockIdx.y;

    const int np = blockIdx.x * 64 + wave * 16 + (lane & 15);
    const int b  = np / HW;          // power-of-two -> shift
    const int rp = np - b * HW;
    const int py = rp / W;
    const int px = rp - py * W;
    const int khalf = (lane >> 4) * 16;

    // base at top-left neighbor (halo coords): padded (py, px) == unpadded (py-1, px-1)
    const half_t* bptr = inslot + (((long)b * Hp + py) * Wp + px) * Kcpad + khalf;
    const half_t* aptr = wpack + (long)mb * 9 * KB * 512 + (long)lane * 16;

    v8f acc = {0.f, 0.f, 0.f, 0.f, 0.f, 0.f, 0.f, 0.f};
#pragma unroll
    for (int tap = 0; tap < 9; ++tap) {
#pragma unroll
        for (int kb = 0; kb < KB; ++kb) {
            v16h bf = *(const v16h*)(bptr + ((tap / 3) * Wp + (tap % 3)) * Kcpad + kb * 32);
            v16h af = *(const v16h*)(aptr + (tap * KB + kb) * 512);
            acc = __builtin_amdgcn_wmma_f32_16x16x32_f16(false, af, false, bf,
                                                         (short)0, acc, false, false);
        }
    }
    // epilogue: C/D layout -> (channel, pixel)
    const int mofs = (lane >> 4) * 8;
#pragma unroll
    for (int r = 0; r < 8; ++r) {
        int ch = mb * 16 + mofs + r;
        z[(((long)b * 4 * Hid + ch) * H + py) * W + px] = acc[r] + bias[ch];
    }
}

// ---------------- LSTM elementwise gate update ----------------
// Writes h_t twice: channel-first hseq slot (for BN/pool) and channel-last into
// the interior of the next conv input slot (if any).
__global__ void lstm_gate(const float* __restrict__ z, float* __restrict__ c,
                          half_t* __restrict__ hout_cf, half_t* __restrict__ hnext_cl,
                          int Hid, int H, int W, int Cin, int Kcpad)
{
    int HW = H * W;
    int i = blockIdx.x * blockDim.x + threadIdx.x;
    int n = B_ * Hid * HW;
    if (i >= n) return;
    int b  = i / (Hid * HW);
    int r  = i - b * Hid * HW;
    int ch = r / HW;
    int p  = r - ch * HW;
    int py = p / W, px = p - py * W;
    long base = (long)b * 4 * Hid * HW + p;
    float zi = z[base + (long)ch * HW];
    float zf = z[base + (long)(Hid + ch) * HW];
    float zo = z[base + (long)(2 * Hid + ch) * HW];
    float zg = z[base + (long)(3 * Hid + ch) * HW];
    float si = 1.f / (1.f + __expf(-zi));
    float sf = 1.f / (1.f + __expf(-zf));
    float so = 1.f / (1.f + __expf(-zo));
    float cn = sf * c[i] + si * tanhf(zg);
    float hn = so * tanhf(cn);
    c[i] = cn;
    half_t h16 = (half_t)hn;
    hout_cf[i] = h16;
    if (hnext_cl)
        hnext_cl[(((long)b * (H + 2) + (py + 1)) * (W + 2) + (px + 1)) * Kcpad + Cin + ch] = h16;
}

// ---------------- BN stats (one block per channel) ----------------
__global__ void bn_stats(const half_t* __restrict__ hseq, float* __restrict__ mean,
                         float* __restrict__ rstd, int C, int HW)
{
    int c = blockIdx.x;
    long N = (long)T_ * B_ * HW;
    float s = 0.f, ss = 0.f;
    for (long i = threadIdx.x; i < N; i += blockDim.x) {
        long tb = i / HW;
        long p  = i - tb * HW;
        float v = (float)hseq[(tb * C + c) * HW + p];
        s += v; ss += v * v;
    }
    __shared__ float sh[256], sh2[256];
    sh[threadIdx.x] = s; sh2[threadIdx.x] = ss;
    __syncthreads();
    for (int o = 128; o > 0; o >>= 1) {
        if ((int)threadIdx.x < o) {
            sh[threadIdx.x]  += sh[threadIdx.x + o];
            sh2[threadIdx.x] += sh2[threadIdx.x + o];
        }
        __syncthreads();
    }
    if (threadIdx.x == 0) {
        float mu  = sh[0] / (float)N;
        float var = sh2[0] / (float)N - mu * mu;
        mean[c] = mu;
        rstd[c] = rsqrtf(var + 1e-5f);
    }
}

// ---------------- BN + ReLU + 2x2 maxpool -> next layer input (channel-first) ----------------
// out layout [b][t][c][H/2][W/2]
__global__ void bn_relu_pool(const half_t* __restrict__ hseq,
                             const float* __restrict__ mean, const float* __restrict__ rstd,
                             const float* __restrict__ gamma, const float* __restrict__ beta,
                             half_t* __restrict__ out, int C, int H, int W)
{
    int H2 = H >> 1, W2 = W >> 1;
    long n = (long)B_ * T_ * C * H2 * W2;
    long i = (long)blockIdx.x * blockDim.x + threadIdx.x;
    if (i >= n) return;
    int x2 = (int)(i % W2); long r = i / W2;
    int y2 = (int)(r % H2); r /= H2;
    int c  = (int)(r % C);  r /= C;
    int t  = (int)(r % T_);
    int b  = (int)(r / T_);
    float mu = mean[c], rs = rstd[c], g = gamma[c], be = beta[c];
    long basein = (((long)t * B_ + b) * C + c) * H * W;
    float best = -3.4e38f;
#pragma unroll
    for (int dy = 0; dy < 2; ++dy)
#pragma unroll
        for (int dx = 0; dx < 2; ++dx) {
            float v = (float)hseq[basein + (2 * y2 + dy) * W + (2 * x2 + dx)];
            v = (v - mu) * rs * g + be;
            if (v > best) best = v;
        }
    best = fmaxf(best, 0.f);
    out[((((long)b * T_ + t) * C + c) * H2 + y2) * W2 + x2] = (half_t)best;
}

// ---------------- FC1: [320 x 1024] x [1024 x 128] WMMA GEMM + bias + ReLU ----------------
__global__ __launch_bounds__(32)
void fc1_wmma(const half_t* __restrict__ feat, const half_t* __restrict__ wpack,
              const float* __restrict__ bias, float* __restrict__ out)
{
    const int lane = threadIdx.x & 31;
    const int nb = blockIdx.x, mb = blockIdx.y;
    v8f acc = {0.f, 0.f, 0.f, 0.f, 0.f, 0.f, 0.f, 0.f};
    const int n = nb * 16 + (lane & 15);
    const int khalf = (lane >> 4) * 16;
#pragma unroll
    for (int kb = 0; kb < 32; ++kb) {
        v16h af = *(const v16h*)&wpack[(((long)mb * 32 + kb) * 32 + lane) * 16];
        v16h bf = *(const v16h*)&feat[(long)n * 1024 + kb * 32 + khalf];
        acc = __builtin_amdgcn_wmma_f32_16x16x32_f16(false, af, false, bf,
                                                     (short)0, acc, false, false);
    }
    const int mofs = (lane >> 4) * 8;
#pragma unroll
    for (int r = 0; r < 8; ++r) {
        int m = mb * 16 + mofs + r;
        out[(long)n * 128 + m] = fmaxf(acc[r] + bias[m], 0.f);
    }
}

// ---------------- FC2: 128 -> 2 ----------------
__global__ void fc2_kernel(const float* __restrict__ f1, const float* __restrict__ w,
                           const float* __restrict__ b, float* __restrict__ y)
{
    int i = blockIdx.x * blockDim.x + threadIdx.x;
    if (i >= 320 * 2) return;
    int n = i >> 1, o = i & 1;
    float s = b[o];
    for (int k = 0; k < 128; ++k) s += f1[n * 128 + k] * w[o * 128 + k];
    y[i] = s;
}

// ---------------- host orchestration ----------------
extern "C" void kernel_launch(void* const* d_in, const int* in_sizes, int n_in,
                              void* d_out, int out_size, void* d_ws, size_t ws_size,
                              hipStream_t stream)
{
    const float* x = (const float*)d_in[0];
    const float* w[4]     = {(const float*)d_in[1],  (const float*)d_in[5],
                             (const float*)d_in[9],  (const float*)d_in[13]};
    const float* bias[4]  = {(const float*)d_in[2],  (const float*)d_in[6],
                             (const float*)d_in[10], (const float*)d_in[14]};
    const float* gamma[4] = {(const float*)d_in[3],  (const float*)d_in[7],
                             (const float*)d_in[11], (const float*)d_in[15]};
    const float* beta[4]  = {(const float*)d_in[4],  (const float*)d_in[8],
                             (const float*)d_in[12], (const float*)d_in[16]};
    const float* fc1w = (const float*)d_in[17];
    const float* fc1b = (const float*)d_in[18];
    const float* fc2w = (const float*)d_in[19];
    const float* fc2b = (const float*)d_in[20];

    const int Cin[4] = {1, 16, 32, 64};
    const int Hid[4] = {16, 32, 64, 64};
    const int Hh[4]  = {64, 32, 16, 8};

    char* ws = (char*)d_ws;
    size_t cur = 0;
    auto alloc = [&](size_t bytes) -> char* {
        char* p = ws + cur;
        cur += (bytes + 255) & ~(size_t)255;
        return p;
    };

    half_t* x16 = (half_t*)alloc((size_t)B_ * T_ * 64 * 64 * 2);
    half_t* wp[4];
    int MBs[4], KBs[4];
    for (int l = 0; l < 4; ++l) {
        int Kc = Cin[l] + Hid[l];
        KBs[l] = (Kc + 31) / 32;
        MBs[l] = (4 * Hid[l]) / 16;
        wp[l] = (half_t*)alloc((size_t)MBs[l] * 9 * KBs[l] * 512 * 2);
    }
    half_t* wpfc  = (half_t*)alloc((size_t)8 * 32 * 512 * 2);
    // channel-last spatially padded conv input, max = layer1: 40*8*66*66*32 halfs (~89 MB)
    half_t* inbuf = (half_t*)alloc((size_t)T_ * B_ * 66 * 66 * 32 * 2);
    half_t* hseq  = (half_t*)alloc((size_t)T_ * B_ * 16 * 64 * 64 * 2); // max over layers
    half_t* poolA = (half_t*)alloc((size_t)B_ * T_ * 16 * 32 * 32 * 2); // max pooled
    half_t* poolB = (half_t*)alloc((size_t)B_ * T_ * 16 * 32 * 32 * 2);
    float*  zbuf  = (float*)alloc((size_t)B_ * 64 * 64 * 64 * 4);       // max 4*Hid*H*W
    float*  cbuf  = (float*)alloc((size_t)B_ * 16 * 64 * 64 * 4);       // max Hid*H*W
    float*  meanb = (float*)alloc(64 * 4);
    float*  rstdb = (float*)alloc(64 * 4);
    float*  f1    = (float*)alloc((size_t)320 * 128 * 4);

    // cast input to f16
    int nx = B_ * T_ * 64 * 64;
    cast_f32_to_f16<<<(nx + 255) / 256, 256, 0, stream>>>(x, x16, nx);

    // pack conv + fc1 weights into WMMA A-fragment layout
    for (int l = 0; l < 4; ++l) {
        int Kc = Cin[l] + Hid[l];
        int nth = MBs[l] * 9 * KBs[l] * 32;
        pack_a_kernel<<<(nth + 255) / 256, 256, 0, stream>>>(
            w[l], wp[l], MBs[l], 9, KBs[l], Kc, Kc * 9, 9);
    }
    {
        int nth = 8 * 1 * 32 * 32;
        pack_a_kernel<<<(nth + 255) / 256, 256, 0, stream>>>(
            fc1w, wpfc, 8, 1, 32, 1024, 1024, 1);
    }

    // layers
    const half_t* lin = x16;
    half_t* pool_out = poolA;
    for (int l = 0; l < 4; ++l) {
        int C = Cin[l], Hd = Hid[l], H = Hh[l], W = Hh[l], HW = H * W;
        int KB = KBs[l], Kcpad = KB * 32;
        int nstate = B_ * Hd * HW;
        size_t slot = (size_t)B_ * (H + 2) * (W + 2) * Kcpad;

        fill_zero_f32<<<(nstate + 255) / 256, 256, 0, stream>>>(cbuf, nstate);
        long ninit = (long)T_ * slot;
        init_inbuf<<<(int)((ninit + 255) / 256), 256, 0, stream>>>(
            lin, inbuf, C, Kcpad, H, W, T_ * C * HW, ninit);

        dim3 grid(B_ * HW / 64, (4 * Hd) / 16);
        for (int t = 0; t < T_; ++t) {
            const half_t* slotp = inbuf + (size_t)t * slot;
            switch (l) {
            case 0: conv_gates_wmma<1, 64, 64><<<grid, 128, 0, stream>>>(slotp, wp[l], bias[l], zbuf, Hd); break;
            case 1: conv_gates_wmma<2, 32, 32><<<grid, 128, 0, stream>>>(slotp, wp[l], bias[l], zbuf, Hd); break;
            case 2: conv_gates_wmma<3, 16, 16><<<grid, 128, 0, stream>>>(slotp, wp[l], bias[l], zbuf, Hd); break;
            default: conv_gates_wmma<4, 8, 8><<<grid, 128, 0, stream>>>(slotp, wp[l], bias[l], zbuf, Hd); break;
            }
            half_t* hnext = (t + 1 < T_) ? inbuf + (size_t)(t + 1) * slot : (half_t*)nullptr;
            lstm_gate<<<(nstate + 255) / 256, 256, 0, stream>>>(
                zbuf, cbuf, hseq + (size_t)t * nstate, hnext, Hd, H, W, C, Kcpad);
        }
        bn_stats<<<Hd, 256, 0, stream>>>(hseq, meanb, rstdb, Hd, HW);
        long npool = (long)B_ * T_ * Hd * (H / 2) * (W / 2);
        bn_relu_pool<<<(int)((npool + 255) / 256), 256, 0, stream>>>(
            hseq, meanb, rstdb, gamma[l], beta[l], pool_out, Hd, H, W);
        lin = pool_out;
        pool_out = (pool_out == poolA) ? poolB : poolA;
    }

    // FC head: feat = [320][1024] f16 (contiguous C,H,W per (b,t))
    fc1_wmma<<<dim3(20, 8), 32, 0, stream>>>(lin, wpfc, fc1b, f1);
    fc2_kernel<<<(640 + 255) / 256, 256, 0, stream>>>(f1, fc2w, fc2b, (float*)d_out);
    (void)in_sizes; (void)n_in; (void)out_size; (void)ws_size;
}
